// BigBirdAttention_56573309223863
// MI455X (gfx1250) — matrix operand order repaired and checked by reading, not compiled, not run on hardware
//
#include <hip/hip_runtime.h>

#define EMB   1024
#define HEADS 16
#define HD    64
#define BATCH 2
#define SEQ   2048
#define MROWS (BATCH * SEQ)   // 4096

#define BM 128
#define BN 128
#define BK 32
#define APITCH 40   // bf16 elements per LDS row (80B): 16B-aligned b128 reads, conflict-minimal
#define BPITCH 40

typedef __attribute__((ext_vector_type(16))) __bf16 bf16x16;
typedef __attribute__((ext_vector_type(2)))  __bf16 bf16x2;
typedef __attribute__((ext_vector_type(2)))  float  f32x2;
typedef __attribute__((ext_vector_type(8)))  float  f32x8;

union FragAB { bf16x16 v; uint4 q[2]; };

// packed f32x2 -> bf16x2 (RNE); lowers to v_cvt_pk_bf16_f32 when available
__device__ __forceinline__ unsigned int pack_bf16(float lo, float hi) {
  f32x2 f; f[0] = lo; f[1] = hi;
  bf16x2 b = __builtin_convertvector(f, bf16x2);
  return __builtin_bit_cast(unsigned int, b);
}

// C[M,EMB] = (A[M,EMB] @ W[EMB,EMB]^T + bias) * alpha   (torch Linear, NT GEMM)
// bf16 WMMA, f32 accumulation, double-buffered LDS.
__global__ __launch_bounds__(256) void gemm_nt_bias(
    const float* __restrict__ A, const float* __restrict__ W,
    const float* __restrict__ bias, float* __restrict__ C, float alpha)
{
  __shared__ __align__(16) unsigned short As[2][BM * APITCH]; // 2 x 10 KB
  __shared__ __align__(16) unsigned short Bs[2][BN * BPITCH]; // 2 x 10 KB
  const int t    = threadIdx.x;
  const int lane = t & 31;
  const int wave = t >> 5;
  const int wm   = wave & 3;      // 4 wave-rows of 32
  const int wn   = wave >> 2;     // 2 wave-cols of 64
  const int mb   = blockIdx.y * BM;
  const int nb   = blockIdx.x * BN;
  const int lh   = lane >> 4;     // half-wave select
  const int l16  = lane & 15;

  f32x8 acc[2][4] = {};

  // Stage one 128x32 A tile + 128x32 B tile (f32 -> bf16) into LDS buffer `buf`.
  // All global loads are issued as one clause before any conversion/ds_store.
  auto stage = [&](int kb, int buf) {
    float4 ra[4], rb[4];
#pragma unroll
    for (int it = 0; it < 4; ++it) {
      const int g = t + it * 256;            // 1024 float4 groups
      const int row = g >> 3, cg = g & 7;
      ra[it] = *reinterpret_cast<const float4*>(
          &A[(size_t)(mb + row) * EMB + kb + cg * 4]);
    }
#pragma unroll
    for (int it = 0; it < 4; ++it) {
      const int g = t + it * 256;
      const int row = g >> 3, cg = g & 7;
      rb[it] = *reinterpret_cast<const float4*>(
          &W[(size_t)(nb + row) * EMB + kb + cg * 4]);
    }
    // branch-free prefetch of the tile after next
    const int kpf = (kb + BK < EMB) ? (kb + BK) : kb;
    {
      const int row = t >> 3, cg = t & 7;
      __builtin_prefetch(&A[(size_t)(mb + row) * EMB + kpf + cg * 4], 0, 1);
      __builtin_prefetch(&W[(size_t)(nb + row) * EMB + kpf + cg * 4], 0, 1);
    }
#pragma unroll
    for (int it = 0; it < 4; ++it) {
      const int g = t + it * 256;
      const int row = g >> 3, cg = g & 7;
      uint2 p;
      p.x = pack_bf16(ra[it].x, ra[it].y);
      p.y = pack_bf16(ra[it].z, ra[it].w);
      *reinterpret_cast<uint2*>(&As[buf][row * APITCH + cg * 4]) = p;
    }
#pragma unroll
    for (int it = 0; it < 4; ++it) {
      const int g = t + it * 256;
      const int row = g >> 3, cg = g & 7;
      uint2 p;
      p.x = pack_bf16(rb[it].x, rb[it].y);
      p.y = pack_bf16(rb[it].z, rb[it].w);
      *reinterpret_cast<uint2*>(&Bs[buf][row * BPITCH + cg * 4]) = p;
    }
  };

  stage(0, 0);
  int buf = 0;
  for (int kb = 0; kb < EMB; kb += BK) {
    __syncthreads();                         // stage(buf) visible; prev reads of buf^1 done
    if (kb + BK < EMB) stage(kb + BK, buf ^ 1);

    // A 16x32 bf16 frag: lane<16 -> K in [lh*8,lh*8+8) and [16+lh*8, ...)
    FragAB a[2], b[4];
#pragma unroll
    for (int tm = 0; tm < 2; ++tm) {
      const int mr = wm * 32 + tm * 16 + l16;
      a[tm].q[0] = *reinterpret_cast<const uint4*>(&As[buf][mr * APITCH + lh * 8]);
      a[tm].q[1] = *reinterpret_cast<const uint4*>(&As[buf][mr * APITCH + 16 + lh * 8]);
    }
    // B 32x16 bf16 frag: column = l16, K in [lh*8,lh*8+8) and [16+lh*8, ...)
#pragma unroll
    for (int tn = 0; tn < 4; ++tn) {
      const int nr = wn * 64 + tn * 16 + l16;
      b[tn].q[0] = *reinterpret_cast<const uint4*>(&Bs[buf][nr * BPITCH + lh * 8]);
      b[tn].q[1] = *reinterpret_cast<const uint4*>(&Bs[buf][nr * BPITCH + 16 + lh * 8]);
    }
#pragma unroll
    for (int tm = 0; tm < 2; ++tm)
#pragma unroll
      for (int tn = 0; tn < 4; ++tn)
        acc[tm][tn] = __builtin_amdgcn_wmma_f32_16x16x32_bf16(
            false, a[tm].v, false, b[tn].v, (short)0, acc[tm][tn], false, false);
    buf ^= 1;
  }

  // Epilogue: D layout VGPR r -> M = r + 8*lh, N = l16
#pragma unroll
  for (int tn = 0; tn < 4; ++tn) {
    const int n  = nb + wn * 64 + tn * 16 + l16;
    const float bv = bias[n];
#pragma unroll
    for (int tm = 0; tm < 2; ++tm) {
      const int mbase = mb + wm * 32 + tm * 16 + lh * 8;
#pragma unroll
      for (int r = 0; r < 8; ++r)
        C[(size_t)(mbase + r) * EMB + n] = (acc[tm][tn][r] + bv) * alpha;
    }
  }
}

// Sparse BigBird rows (i >= 2): one wave per (b,h,i); <=12 candidate columns with
// exact dedup; scale 1/sqrt(D) already folded into Q.
__global__ __launch_bounds__(256) void bigbird_sparse(
    const float* __restrict__ Q, const float* __restrict__ K,
    const float* __restrict__ V, const int* __restrict__ rnd,
    float* __restrict__ O)
{
  __shared__ float s_p[8][16];
  __shared__ int   s_c[8][16];
  const int lane = threadIdx.x & 31;
  const int wave = threadIdx.x >> 5;
  const int flat = blockIdx.x * 8 + wave;          // b*H*S + h*S + i
  const int i  = flat & (SEQ - 1);
  const int bh = flat >> 11;
  const int h  = bh & (HEADS - 1);
  const int b  = bh >> 4;
  if (i < 2) return;                                // dense rows handled elsewhere

  int c = 0; bool valid = false;
  if (lane < 7) {                                   // window i-3..i+3, drop cols<2 (global slots)
    c = i - 3 + lane;
    valid = (c >= 2) && (c < SEQ);
  } else if (lane < 9) {                            // global cols 0,1
    c = lane - 7;
    valid = true;
  } else if (lane < 12) {                           // randoms, deduped
    const int r0 = rnd[i * 3 + 0], r1 = rnd[i * 3 + 1], r2 = rnd[i * 3 + 2];
    const int rr = (lane == 9) ? r0 : ((lane == 10) ? r1 : r2);
    valid = (rr >= 2) && (rr < i - 3 || rr > i + 3);
    if (lane == 10) valid = valid && (rr != r0);
    if (lane == 11) valid = valid && (rr != r0) && (rr != r1);
    c = rr < 0 ? 0 : rr;
  }

  const float* qp = Q + ((size_t)(b * SEQ + i)) * EMB + h * HD;
  const float* kp = K + ((size_t)(b * SEQ + c)) * EMB + h * HD;
  float s = 0.f;
#pragma unroll
  for (int d4 = 0; d4 < HD / 4; ++d4) {
    const float4 qv = *reinterpret_cast<const float4*>(qp + d4 * 4);
    const float4 kv = *reinterpret_cast<const float4*>(kp + d4 * 4);
    s += qv.x * kv.x + qv.y * kv.y + qv.z * kv.z + qv.w * kv.w;
  }
  if (!valid) s = -__builtin_inff();

  float m = s;
#pragma unroll
  for (int off = 16; off >= 1; off >>= 1) m = fmaxf(m, __shfl_xor(m, off, 32));
  const float e = valid ? __expf(s - m) : 0.f;
  float l = e;
#pragma unroll
  for (int off = 16; off >= 1; off >>= 1) l += __shfl_xor(l, off, 32);
  const float p = e / l;

  if (lane < 16) {                // publish probs + columns (same-wave LDS: in order)
    s_p[wave][lane] = (lane < 12) ? p : 0.f;
    s_c[wave][lane] = (lane < 12) ? c : 0;
  }

  // d-parallel accumulate: each lane owns dims 2*lane, 2*lane+1
  float2 acc; acc.x = 0.f; acc.y = 0.f;
#pragma unroll 1
  for (int tt = 0; tt < 12; ++tt) {
    const float pt = s_p[wave][tt];
    const int   ct = s_c[wave][tt];
    const float2 vv = *reinterpret_cast<const float2*>(
        V + ((size_t)(b * SEQ + ct)) * EMB + h * HD + 2 * lane);
    acc.x += pt * vv.x;
    acc.y += pt * vv.y;
  }
  *reinterpret_cast<float2*>(O + ((size_t)(b * SEQ + i)) * EMB + h * HD + 2 * lane) = acc;
}

// Dense rows i < 2: full 2048-column softmax; one wave per (b,h,i), scores in LDS.
__global__ __launch_bounds__(32) void bigbird_global(
    const float* __restrict__ Q, const float* __restrict__ K,
    const float* __restrict__ V, float* __restrict__ O)
{
  __shared__ __align__(16) float sc[SEQ];
  __shared__ __align__(16) float qs[HD];
  const int lane = threadIdx.x;
  const int flat = blockIdx.x;                 // b*32 + h*2 + i
  const int i = flat & 1;
  const int h = (flat >> 1) & (HEADS - 1);
  const int b = flat >> 5;

  const float* qp = Q + ((size_t)(b * SEQ + i)) * EMB + h * HD;
  qs[lane] = qp[lane];
  qs[lane + 32] = qp[lane + 32];
  __syncthreads();

  float m = -__builtin_inff();
  for (int j = lane; j < SEQ; j += 32) {
    const float* kp = K + ((size_t)(b * SEQ + j)) * EMB + h * HD;
    float s = 0.f;
#pragma unroll
    for (int d4 = 0; d4 < HD / 4; ++d4) {
      const float4 qv = *reinterpret_cast<const float4*>(qs + d4 * 4);
      const float4 kv = *reinterpret_cast<const float4*>(kp + d4 * 4);
      s += qv.x * kv.x + qv.y * kv.y + qv.z * kv.z + qv.w * kv.w;
    }
    sc[j] = s;
    m = fmaxf(m, s);
  }
#pragma unroll
  for (int off = 16; off >= 1; off >>= 1) m = fmaxf(m, __shfl_xor(m, off, 32));
  float l = 0.f;
  for (int j = lane; j < SEQ; j += 32) l += __expf(sc[j] - m);
#pragma unroll
  for (int off = 16; off >= 1; off >>= 1) l += __shfl_xor(l, off, 32);
  const float inv = 1.f / l;

  float2 acc; acc.x = 0.f; acc.y = 0.f;
  for (int j = 0; j < SEQ; ++j) {
    const float p = __expf(sc[j] - m) * inv;   // broadcast LDS read
    const float2 vv = *reinterpret_cast<const float2*>(
        V + ((size_t)(b * SEQ + j)) * EMB + h * HD + 2 * lane);
    acc.x += p * vv.x;
    acc.y += p * vv.y;
  }
  *reinterpret_cast<float2*>(O + ((size_t)(b * SEQ + i)) * EMB + h * HD + 2 * lane) = acc;
}

extern "C" void kernel_launch(void* const* d_in, const int* in_sizes, int n_in,
                              void* d_out, int out_size, void* d_ws, size_t ws_size,
                              hipStream_t stream) {
  const float* x   = (const float*)d_in[0];
  const int*   rnd = (const int*)  d_in[1];
  const float* q_w = (const float*)d_in[2];
  const float* q_b = (const float*)d_in[3];
  const float* k_w = (const float*)d_in[4];
  const float* k_b = (const float*)d_in[5];
  const float* v_w = (const float*)d_in[6];
  const float* v_b = (const float*)d_in[7];
  const float* o_w = (const float*)d_in[8];
  const float* o_b = (const float*)d_in[9];
  float* out = (float*)d_out;

  float* ws = (float*)d_ws;
  const size_t per = (size_t)MROWS * EMB;      // 16 MB each
  float* Qf = ws;
  float* Kf = ws + per;
  float* Vf = ws + 2 * per;
  float* At = ws + 3 * per;

  dim3 gg(EMB / BN, MROWS / BM);               // 8 x 32 blocks
  // Q projection: fold 1/sqrt(HD)=0.125 into Q (scales q.k exactly as reference)
  gemm_nt_bias<<<gg, 256, 0, stream>>>(x, q_w, q_b, Qf, 0.125f);
  gemm_nt_bias<<<gg, 256, 0, stream>>>(x, k_w, k_b, Kf, 1.0f);
  gemm_nt_bias<<<gg, 256, 0, stream>>>(x, v_w, v_b, Vf, 1.0f);

  bigbird_sparse<<<(BATCH * HEADS * SEQ) / 8, 256, 0, stream>>>(Qf, Kf, Vf, rnd, At);
  bigbird_global<<<BATCH * HEADS * 2, 32, 0, stream>>>(Qf, Kf, Vf, At);

  gemm_nt_bias<<<gg, 256, 0, stream>>>(At, o_w, o_b, out, 1.0f);
}
